// Model_39676907885287
// MI455X (gfx1250) — matrix operand order repaired and checked by reading, not compiled, not run on hardware
//
#include <hip/hip_runtime.h>

typedef __attribute__((ext_vector_type(16))) _Float16 v16h;
typedef __attribute__((ext_vector_type(8)))  _Float16 v8h;
typedef __attribute__((ext_vector_type(2)))  _Float16 v2h;
typedef __attribute__((ext_vector_type(8)))  float    v8f;
typedef __attribute__((ext_vector_type(4)))  int      v4i;

static constexpr int kBatch = 16;
static constexpr int kSeq   = 2048;
static constexpr int kDim   = 64;
static constexpr int kKB    = 128;             // K/V rows per block iteration (softmax span)
static constexpr int kNB    = kKB / 16;        // 8 S-tiles per block
static constexpr int kQW    = 16;              // Q rows per wave
static constexpr int kWaves = 4;               // 128 threads, wave32
static constexpr int kQB    = kQW * kWaves;    // 64 Q rows per workgroup
static constexpr int kVT    = kKB + 8;         // padded stride (halves) for V^T in LDS

// fold softmax scale (1/sqrt(64)) and log2(e) into Q so exp() becomes raw v_exp_f32
#define QK_PRESCALE (0.125f * 1.44269504088896340736f)

// ---- CDNA5 async global->LDS path (guarded; falls back to VGPR-staged copies) ----
#if __has_builtin(__builtin_amdgcn_global_load_async_to_lds_b128) && \
    __has_builtin(__builtin_amdgcn_s_wait_asynccnt)
#define USE_ASYNC_LDS 1
typedef __attribute__((address_space(1))) v4i* gas_ptr;   // global b128
typedef __attribute__((address_space(3))) v4i* las_ptr;   // LDS b128
#else
#define USE_ASYNC_LDS 0
#endif

// ---- DPP butterfly reduction within each 16-lane half (wave32) ----
template <int CTRL>
__device__ __forceinline__ float dppf(float v) {
  return __builtin_bit_cast(float,
      __builtin_amdgcn_update_dpp(0, __builtin_bit_cast(int, v), CTRL, 0xF, 0xF, false));
}
__device__ __forceinline__ float rowmax16(float v) {
  v = fmaxf(v, dppf<0xB1>(v));    // quad_perm [1,0,3,2]
  v = fmaxf(v, dppf<0x4E>(v));    // quad_perm [2,3,0,1]
  v = fmaxf(v, dppf<0x141>(v));   // row_half_mirror
  v = fmaxf(v, dppf<0x140>(v));   // row_mirror
  return v;
}
__device__ __forceinline__ float rowsum16(float v) {
  v += dppf<0xB1>(v);
  v += dppf<0x4E>(v);
  v += dppf<0x141>(v);
  v += dppf<0x140>(v);
  return v;
}
__device__ __forceinline__ v16h cat8(v8h lo, v8h hi) {
  return __builtin_shufflevector(lo, hi, 0, 1, 2, 3, 4, 5, 6, 7,
                                 8, 9, 10, 11, 12, 13, 14, 15);
}

// ---- one-time pre-pass: x2 (fp32) -> f16, row-major K16[b][s][d] and Vt16[b][d][s] ----
__global__ __launch_bounds__(256) void cvt_kernel(const float* __restrict__ x2,
                                                  _Float16* __restrict__ K16,
                                                  _Float16* __restrict__ Vt16) {
  const int gid = blockIdx.x * 256 + threadIdx.x;   // 262144 threads
  {
    const size_t base = (size_t)gid * 8;            // 8 contiguous elems, row-major
    float4 a = *(const float4*)(x2 + base);
    float4 c = *(const float4*)(x2 + base + 4);
    *(v8h*)(K16 + base) = (v8h){(_Float16)a.x, (_Float16)a.y, (_Float16)a.z, (_Float16)a.w,
                                (_Float16)c.x, (_Float16)c.y, (_Float16)c.z, (_Float16)c.w};
  }
  {
    const int b = gid >> 14, d = (gid >> 8) & 63, s0 = (gid & 255) * 8;
    const float* src = x2 + ((size_t)b * kSeq + s0) * kDim + d;
    v8h o;
#pragma unroll
    for (int j = 0; j < 8; ++j) o[j] = (_Float16)src[(size_t)j * kDim];
    *(v8h*)(Vt16 + ((size_t)b * kDim + d) * kSeq + s0) = o;
  }
}

template <bool PRE>
__global__ __launch_bounds__(128) void fa_fwd_kernel(const float* __restrict__ x1,
                                                     const float* __restrict__ x2,
                                                     const _Float16* __restrict__ K16,
                                                     const _Float16* __restrict__ Vt16,
                                                     float* __restrict__ out) {
  __shared__ __align__(16) _Float16 ldsK[kKB * kDim];          // K block, row-major (f16)
  __shared__ __align__(16) _Float16 ldsVt[kDim * kVT];         // V block transposed (f16)
  __shared__ __align__(16) _Float16 ldsP[kWaves][kQW * kKB];   // per-wave P tile (f16)

  const int tid  = threadIdx.x;
  const int lane = tid & 31;
  const int wid  = tid >> 5;
  const int ln16 = lane & 15;
  const int h    = lane >> 4;   // half-wave select (0/1)

  const int b  = blockIdx.x >> 5;                 // 32 q-blocks per batch
  const int q0 = (blockIdx.x & 31) * kQB + wid * kQW;

  // ---- Q tile (16x64 fp32) -> two pre-scaled f16 A-fragments (16x16x32 layout) ----
  const float* qrow = x1 + ((size_t)b * kSeq + (size_t)(q0 + ln16)) * kDim;
  v16h qa[2];
#pragma unroll
  for (int kf = 0; kf < 2; ++kf) {
    const int d0 = kf * 32 + h * 16;
    v16h a;
#pragma unroll
    for (int j = 0; j < 8; ++j) {
      a[j]     = (_Float16)(qrow[d0 + j]      * QK_PRESCALE);
      a[j + 8] = (_Float16)(qrow[d0 + 8 + j] * QK_PRESCALE);
    }
    qa[kf] = a;
  }

  v8f oacc[4] = {};
  float run_m[8], run_l[8];
#pragma unroll
  for (int i = 0; i < 8; ++i) { run_m[i] = -3.0e38f; run_l[i] = 0.0f; }

  _Float16* pb = ldsP[wid];

  for (int kb = 0; kb < kSeq / kKB; ++kb) {
    __syncthreads();   // previous block's LDS reads complete before overwrite

    if constexpr (PRE) {
      const _Float16* ksrc = K16 + ((size_t)b * kSeq + (size_t)(kb * kKB + tid)) * kDim;
      _Float16* kdst = &ldsK[tid * kDim];
      const int vrow = tid >> 1, vseg = tid & 1;
      const _Float16* vsrc = Vt16 + ((size_t)b * kDim + vrow) * kSeq
                             + (size_t)(kb * kKB + vseg * 64);
      _Float16* vdst = &ldsVt[vrow * kVT + vseg * 64];
#if USE_ASYNC_LDS
      // ---- CDNA5 async DMA: global -> LDS, no VGPR staging (ASYNCcnt) ----
#pragma unroll
      for (int j = 0; j < 8; ++j)
        __builtin_amdgcn_global_load_async_to_lds_b128(
            (gas_ptr)(ksrc + j * 8), (las_ptr)(kdst + j * 8), 0, 0);
#pragma unroll
      for (int j = 0; j < 8; ++j)
        __builtin_amdgcn_global_load_async_to_lds_b128(
            (gas_ptr)(vsrc + j * 8), (las_ptr)(vdst + j * 8), 0, 0);
      __builtin_amdgcn_s_wait_asynccnt(0);
#else
      const uint4* ks = (const uint4*)ksrc;
      uint4* kd = (uint4*)kdst;
#pragma unroll
      for (int j = 0; j < 8; ++j) kd[j] = ks[j];
      const uint4* vs = (const uint4*)vsrc;
      uint4* vd = (uint4*)vdst;
#pragma unroll
      for (int j = 0; j < 8; ++j) vd[j] = vs[j];
#endif
    } else {
      // ---- fallback: convert fp32 K/V block in-loop (scratch too small) ----
      const int r0 = (tid >> 3) * 8;
      const int c0 = (tid & 7) * 8;
      const float* src = x2 + ((size_t)b * kSeq + (size_t)(kb * kKB + r0)) * kDim + c0;
#pragma unroll
      for (int rp = 0; rp < 4; ++rp) {
        const int ra = r0 + 2 * rp, rb = ra + 1;
        float4 f0a = *(const float4*)(src + (size_t)(2 * rp) * kDim);
        float4 f0b = *(const float4*)(src + (size_t)(2 * rp) * kDim + 4);
        float4 f1a = *(const float4*)(src + (size_t)(2 * rp + 1) * kDim);
        float4 f1b = *(const float4*)(src + (size_t)(2 * rp + 1) * kDim + 4);
        _Float16 ha[8] = {(_Float16)f0a.x, (_Float16)f0a.y, (_Float16)f0a.z, (_Float16)f0a.w,
                          (_Float16)f0b.x, (_Float16)f0b.y, (_Float16)f0b.z, (_Float16)f0b.w};
        _Float16 hb[8] = {(_Float16)f1a.x, (_Float16)f1a.y, (_Float16)f1a.z, (_Float16)f1a.w,
                          (_Float16)f1b.x, (_Float16)f1b.y, (_Float16)f1b.z, (_Float16)f1b.w};
        *(v8h*)&ldsK[ra * kDim + c0] = (v8h){ha[0], ha[1], ha[2], ha[3],
                                             ha[4], ha[5], ha[6], ha[7]};
        *(v8h*)&ldsK[rb * kDim + c0] = (v8h){hb[0], hb[1], hb[2], hb[3],
                                             hb[4], hb[5], hb[6], hb[7]};
#pragma unroll
        for (int j = 0; j < 8; ++j)
          *(v2h*)&ldsVt[(c0 + j) * kVT + ra] = (v2h){ha[j], hb[j]};
      }
    }
    __syncthreads();

    // ---- S = Q * K^T : eight 16x16 tiles; load B-frags in pairs ahead of WMMAs ----
    v8f sfr[kNB];
#pragma unroll
    for (int nb = 0; nb < kNB; ++nb) sfr[nb] = (v8f){};
#pragma unroll
    for (int kf = 0; kf < 2; ++kf) {
#pragma unroll
      for (int ng = 0; ng < kNB; ng += 2) {
        const _Float16* kr0 = &ldsK[((ng + 0) * 16 + ln16) * kDim + kf * 32 + h * 16];
        const _Float16* kr1 = &ldsK[((ng + 1) * 16 + ln16) * kDim + kf * 32 + h * 16];
        v16h bf0 = cat8(*(const v8h*)kr0, *(const v8h*)(kr0 + 8));
        v16h bf1 = cat8(*(const v8h*)kr1, *(const v8h*)(kr1 + 8));
        sfr[ng + 0] = __builtin_amdgcn_wmma_f32_16x16x32_f16(false, qa[kf], false, bf0,
                                                             (short)0, sfr[ng + 0], false, false);
        sfr[ng + 1] = __builtin_amdgcn_wmma_f32_16x16x32_f16(false, qa[kf], false, bf1,
                                                             (short)0, sfr[ng + 1], false, false);
      }
    }

    // ---- online softmax over 128 columns (log2-domain; S pre-scaled) ----
    float bm[8];
    bool grow = false;
#pragma unroll
    for (int i = 0; i < 8; ++i) {
      float t = fmaxf(sfr[0][i], sfr[1][i]);
#pragma unroll
      for (int nb = 2; nb < kNB; ++nb) t = fmaxf(t, sfr[nb][i]);
      bm[i] = rowmax16(t);
      grow = grow || (bm[i] > run_m[i]);
    }
    if (__any(grow)) {
      // slow path: running max changed somewhere in the wave
#pragma unroll
      for (int i = 0; i < 8; ++i) {
        float nm = fmaxf(run_m[i], bm[i]);
        float alpha = __builtin_amdgcn_exp2f(run_m[i] - nm);
        run_m[i] = nm;
        float rs = 0.0f;
#pragma unroll
        for (int nb = 0; nb < kNB; ++nb) {
          float p = __builtin_amdgcn_exp2f(sfr[nb][i] - nm);
          sfr[nb][i] = p;
          rs += p;
        }
        run_l[i] = run_l[i] * alpha + rowsum16(rs);
#pragma unroll
        for (int db = 0; db < 4; ++db) oacc[db][i] *= alpha;
      }
    } else {
      // fast path: max unchanged for every row in the wave; no rescale needed
#pragma unroll
      for (int i = 0; i < 8; ++i) {
        float rs = 0.0f;
#pragma unroll
        for (int nb = 0; nb < kNB; ++nb) {
          float p = __builtin_amdgcn_exp2f(sfr[nb][i] - run_m[i]);
          sfr[nb][i] = p;
          rs += p;
        }
        run_l[i] += rowsum16(rs);
      }
    }

    // ---- stage P (16x128) as f16 in per-wave LDS (C-layout -> row-major) ----
#pragma unroll
    for (int nb = 0; nb < kNB; ++nb)
#pragma unroll
      for (int i = 0; i < 8; ++i)
        pb[(i + 8 * h) * kKB + nb * 16 + ln16] = (_Float16)sfr[nb][i];

    // ---- O += P * V : K-dim 128; all 4 B-frags loaded before the 4 WMMAs ----
#pragma unroll
    for (int kf = 0; kf < 4; ++kf) {
      const _Float16* pr = pb + ln16 * kKB + kf * 32 + h * 16;
      v16h pa = cat8(*(const v8h*)pr, *(const v8h*)(pr + 8));
      v16h bfv[4];
#pragma unroll
      for (int db = 0; db < 4; ++db) {
        const _Float16* vr = &ldsVt[(db * 16 + ln16) * kVT + kf * 32 + h * 16];
        bfv[db] = cat8(*(const v8h*)vr, *(const v8h*)(vr + 8));
      }
#pragma unroll
      for (int db = 0; db < 4; ++db)
        oacc[db] = __builtin_amdgcn_wmma_f32_16x16x32_f16(false, pa, false, bfv[db],
                                                          (short)0, oacc[db], false, false);
    }
  }

  // ---- normalize and store fp32 output ----
  float inv_l[8];
#pragma unroll
  for (int i = 0; i < 8; ++i) inv_l[i] = 1.0f / run_l[i];
  float* orow = out + ((size_t)b * kSeq + (size_t)q0) * kDim;
#pragma unroll
  for (int db = 0; db < 4; ++db)
#pragma unroll
    for (int i = 0; i < 8; ++i)
      orow[(size_t)(i + 8 * h) * kDim + db * 16 + ln16] = oacc[db][i] * inv_l[i];
}

extern "C" void kernel_launch(void* const* d_in, const int* in_sizes, int n_in,
                              void* d_out, int out_size, void* d_ws, size_t ws_size,
                              hipStream_t stream) {
  const float* x1 = (const float*)d_in[0];
  const float* x2 = (const float*)d_in[1];
  float* out = (float*)d_out;
  (void)in_sizes; (void)n_in; (void)out_size;

  const size_t elems = (size_t)kBatch * kSeq * kDim;        // 2.1M
  const size_t need  = elems * 2 * sizeof(_Float16);        // K16 + Vt16 = 8 MB
  dim3 grid(kBatch * (kSeq / kQB));                         // 512 workgroups
  dim3 block(128);                                          // 4 wave32s

  if (ws_size >= need) {
    _Float16* K16  = (_Float16*)d_ws;
    _Float16* Vt16 = K16 + elems;
    hipLaunchKernelGGL(cvt_kernel, dim3(elems / 8 / 256), dim3(256), 0, stream,
                       x2, K16, Vt16);
    hipLaunchKernelGGL((fa_fwd_kernel<true>), grid, block, 0, stream,
                       x1, x2, K16, Vt16, out);
  } else {
    hipLaunchKernelGGL((fa_fwd_kernel<false>), grid, block, 0, stream,
                       x1, x2, (const _Float16*)nullptr, (const _Float16*)nullptr, out);
  }
}
